// Block_22170621182450
// MI455X (gfx1250) — compile-verified, matrix operand order checked by reading, not run on hardware
//
#include <hip/hip_runtime.h>
#include <cstdint>
#include <cstddef>

// Shapes: T=4 B=16 N=256 C=512 H=8 hd=64 HID=2048, M = T*B*N = 16384
// All matmuls run on v_wmma_f32_16x16x32_bf16 (wave32, CDNA5).
// Attention q-tile staging uses the Tensor Data Mover (tensor_load_to_lds).

#define USE_TDM 1

typedef __attribute__((ext_vector_type(16))) __bf16 v16bf;
typedef __attribute__((ext_vector_type(8)))  float  v8f;
typedef __attribute__((ext_vector_type(4)))  unsigned int u32x4;
typedef __attribute__((ext_vector_type(4)))  int i32x4;
typedef __attribute__((ext_vector_type(8)))  int i32x8;

#define BF_WMMA(a, b, c) \
  __builtin_amdgcn_wmma_f32_16x16x32_bf16(false, (a), false, (b), (short)0, (c), false, false)

// ---------------------------------------------------------------------------
// TDM: DMA a 2D tile of 16-bit elements (rows x row_elems, global row stride
// row_stride_elems) into LDS at lds_off, packed [rows][row_elems].
// D# layout per CDNA5 ISA 8.3/8.4 (group0: count/lds/global/type,
// group1: data_size, tensor dims, tile dims, dim0 stride; groups 2/3 zero).
// Issue from ONE wave (uniform branch), then s_wait_tensorcnt + barrier.
// ---------------------------------------------------------------------------
__device__ __forceinline__ void tdm_load_2d_to_lds(const void* gsrc, unsigned lds_off,
                                                   int row_elems, int rows,
                                                   int row_stride_elems) {
  unsigned long long ga = (unsigned long long)gsrc;
  u32x4 g0;
  g0[0] = 1u;                                                    // count=1 (valid)
  g0[1] = lds_off;                                               // LDS byte address
  g0[2] = (unsigned)ga;                                          // global addr lo
  g0[3] = (unsigned)((ga >> 32) & 0x01FFFFFFull) | 0x80000000u;  // addr hi | type=2
  i32x8 g1;
  g1[0] = 0x00010000;                                            // data_size = 2B
  g1[1] = (row_elems & 0xFFFF) << 16;                            // tensor_dim0 lo16
  g1[2] = ((row_elems >> 16) & 0xFFFF) | ((rows & 0xFFFF) << 16); // dim0 hi|dim1 lo
  g1[3] = ((rows >> 16) & 0xFFFF) | ((row_elems & 0xFFFF) << 16); // dim1 hi|tile_dim0
  g1[4] = rows & 0xFFFF;                                         // tile_dim1 (dim2=0)
  g1[5] = row_stride_elems;                                      // dim0_stride lo32
  g1[6] = 0;
  g1[7] = 0;
  i32x4 gz = {0, 0, 0, 0};
#if defined(__clang_major__) && (__clang_major__ >= 23)
  i32x8 gz8 = {0, 0, 0, 0, 0, 0, 0, 0};
  __builtin_amdgcn_tensor_load_to_lds(g0, g1, gz, gz, gz8, 0);
#else
  __builtin_amdgcn_tensor_load_to_lds(g0, g1, gz, gz, 0);
#endif
}

// ---------------------------------------------------------------------------
// Fast fragment loader (CDNA5 ISA 7.12.2, wave32 16-bit 16x32 operand layout):
// two contiguous 16B ranges per lane -> exactly two ds_load_b128.
// ---------------------------------------------------------------------------
__device__ __forceinline__ v16bf load_frag(const __bf16* p, int ld, int lane, int kofs) {
  const int m  = lane & 15;
  const int kb = ((lane >> 4) << 3) + kofs;
  const __bf16* q = p + m * ld + kb;
  v16bf a;
  *(uint4*)&a       = *(const uint4*)(q);        // K = kb .. kb+7
  *((uint4*)&a + 1) = *(const uint4*)(q + 16);   // K = kb+16 .. kb+23
  return a;
}

// ---------------------------------------------------------------------------
// f32 -> bf16 pack (plain)
// ---------------------------------------------------------------------------
__global__ __launch_bounds__(256) void cvt_bf16_kernel(const float* __restrict__ a,
                                                       __bf16* __restrict__ b, size_t n) {
  size_t i = (size_t)blockIdx.x * 256 + threadIdx.x;
  if (i < n) b[i] = (__bf16)a[i];
}

// f32 W[k][n] -> bf16 W^T[n][k]  (one-time; makes GEMM B-tiles K-contiguous)
__global__ __launch_bounds__(256) void cvt_wt_kernel(const float* __restrict__ w,
                                                     __bf16* __restrict__ wt,
                                                     int K, int N) {
  size_t i = (size_t)blockIdx.x * 256 + threadIdx.x;  // over K*N
  if (i >= (size_t)K * N) return;
  int n = (int)(i / K), k = (int)(i % K);
  wt[i] = (__bf16)w[(size_t)k * N + n];
}

// ---------------------------------------------------------------------------
// GEMM: C(MxN) = A(MxK bf16 row-major) * W(KxN) + bias, f32 out.
// W supplied TRANSPOSED (Wt, N-major). Block tile 128x64, 8 waves,
// wave = one 16x64 strip (4 wmma accumulators). LDS B staged column-major.
// ---------------------------------------------------------------------------
__global__ __launch_bounds__(256) void gemm_bf16_kernel(
    const __bf16* __restrict__ A, const __bf16* __restrict__ Wt,
    const float* __restrict__ bias, float* __restrict__ C,
    int M, int K, int Nd) {
  __shared__ __bf16 sA[128 * 32];   // [m][k]
  __shared__ __bf16 sB[64 * 32];    // [n][k]  (column-major tile)
  const int tid = threadIdx.x, lane = tid & 31, wave = tid >> 5;
  const int m0 = blockIdx.x * 128, n0 = blockIdx.y * 64;

  v8f acc[4] = {};

  const int arow = tid >> 1, acol = (tid & 1) << 4;  // 128 rows, 16 bf16/thread
  const int brow = tid >> 2, bcol = (tid & 3) << 3;  // 64 rows,  8 bf16/thread

  for (int k0 = 0; k0 < K; k0 += 32) {
    const __bf16* ag = A + (size_t)(m0 + arow) * K + k0 + acol;
    *(uint4*)&sA[arow * 32 + acol]     = *(const uint4*)(ag);
    *(uint4*)&sA[arow * 32 + acol + 8] = *(const uint4*)(ag + 8);
    *(uint4*)&sB[brow * 32 + bcol] =
        *(const uint4*)(Wt + (size_t)(n0 + brow) * K + k0 + bcol);
    if (k0 + 32 < K) {  // global_prefetch_b8 for next K tile
      __builtin_prefetch(A + (size_t)(m0 + arow) * K + k0 + 32 + acol, 0, 1);
      __builtin_prefetch(Wt + (size_t)(n0 + brow) * K + k0 + 32 + bcol, 0, 1);
    }
    __syncthreads();
    v16bf af = load_frag(&sA[wave * 16 * 32], 32, lane, 0);
#pragma unroll
    for (int nf = 0; nf < 4; ++nf) {
      v16bf bfr = load_frag(&sB[nf * 16 * 32], 32, lane, 0);
      acc[nf] = BF_WMMA(af, bfr, acc[nf]);
    }
    __syncthreads();
  }

  // D layout: VGPR r -> row r (+8 for lanes 16-31), col = lane&15
  const int mb = m0 + wave * 16 + ((lane >> 4) << 3);
  const int nc = lane & 15;
#pragma unroll
  for (int nf = 0; nf < 4; ++nf) {
    int col = n0 + nf * 16 + nc;
    float bv = bias[col];
#pragma unroll
    for (int r = 0; r < 8; ++r)
      C[(size_t)(mb + r) * Nd + col] = acc[nf][r] + bv;
  }
}

// ---------------------------------------------------------------------------
// Fused spiking attention, one (t,b,h, 128-q-row block) per block.
//   attn = (q k^T) * 0.125 ;  o = attn v
// q spikes: (T,B,N,C) row-major (q tile staged via TDM). v spikes:
// PRE-TRANSPOSED (T,B,H,hd,N). k/v streamed in 32-row chunks; per-wave LDS
// scratch re-lays the D-format attn tile into an A-format operand.
// ---------------------------------------------------------------------------
__global__ __launch_bounds__(256) void attn_kernel(
    const __bf16* __restrict__ sq, const __bf16* __restrict__ sk,
    const __bf16* __restrict__ svT, float* __restrict__ O) {
  __shared__ __bf16 lq[128 * 64];   // [qrow][hd]
  __shared__ __bf16 lk[32 * 64];    // [krow][hd]
  __shared__ __bf16 lv[64 * 32];    // [dv][krow]  (column-major v chunk)
  __shared__ __bf16 lattn[8 * 16 * 32];  // per-wave 16x32 scratch

  const int tid = threadIdx.x, lane = tid & 31, wave = tid >> 5;
  const int bid  = blockIdx.x;
  const int qblk = bid & 1;        // 2 q-blocks of 128 rows
  const int tbh  = bid >> 1;       // (t*B+b)*H + h
  const int hh   = tbh & 7;
  const int tb   = tbh >> 3;       // t*B+b in [0,64)
  const size_t base  = (size_t)tb * 256 * 512 + (size_t)hh * 64;   // q,k
  const size_t baseV = (size_t)(tb * 8 + hh) * 64 * 256;           // svT

#if USE_TDM
  if (wave == 0) {  // one TDM per block: 128x64 bf16 tile, row stride 512
    tdm_load_2d_to_lds(sq + base + (size_t)(qblk * 128) * 512,
                       (unsigned)(size_t)&lq[0], 64, 128, 512);
    __builtin_amdgcn_s_wait_tensorcnt(0);
  }
#else
  {  // stage q tile (128x64)
    int row = tid >> 1, col = (tid & 1) * 32;
    const __bf16* g = sq + base + (size_t)(qblk * 128 + row) * 512 + col;
#pragma unroll
    for (int i = 0; i < 4; ++i)
      *(uint4*)&lq[row * 64 + col + i * 8] = *(const uint4*)(g + i * 8);
  }
#endif
  __syncthreads();

  v16bf aq0 = load_frag(&lq[wave * 16 * 64], 64, lane, 0);
  v16bf aq1 = load_frag(&lq[wave * 16 * 64], 64, lane, 32);
  v8f oacc[4] = {};

  const int krow = tid >> 3, kcol = (tid & 7) << 3;   // k chunk: 32x64
  const int vrow = tid >> 2, vcol = (tid & 3) << 3;   // v chunk: 64x32
  for (int kk = 0; kk < 8; ++kk) {  // 32 k-rows per step, 256 total
    __syncthreads();
    *(uint4*)&lk[krow * 64 + kcol] =
        *(const uint4*)(sk + base + (size_t)(kk * 32 + krow) * 512 + kcol);
    *(uint4*)&lv[vrow * 32 + vcol] =
        *(const uint4*)(svT + baseV + (size_t)vrow * 256 + kk * 32 + vcol);
    __syncthreads();

    __bf16* scr = &lattn[wave * 16 * 32];
#pragma unroll
    for (int kt = 0; kt < 2; ++kt) {
      v8f aacc = {};
      // B operand of q*k^T: lane -> attn column (= k row), K = hd contiguous
      v16bf bk0 = load_frag(&lk[kt * 16 * 64], 64, lane, 0);
      v16bf bk1 = load_frag(&lk[kt * 16 * 64], 64, lane, 32);
      aacc = BF_WMMA(aq0, bk0, aacc);
      aacc = BF_WMMA(aq1, bk1, aacc);
      // scale + convert, D-layout -> row-major scratch (same-wave only)
      int rloc = (lane >> 4) << 3;
      int cloc = (lane & 15) + kt * 16;
#pragma unroll
      for (int r = 0; r < 8; ++r)
        scr[(rloc + r) * 32 + cloc] = (__bf16)(aacc[r] * 0.125f);
    }
    v16bf aat = load_frag(scr, 32, lane, 0);  // attn 16x32 as A operand
#pragma unroll
    for (int nf = 0; nf < 4; ++nf) {
      v16bf bv = load_frag(&lv[nf * 16 * 32], 32, lane, 0);  // column-major v
      oacc[nf] = BF_WMMA(aat, bv, oacc[nf]);
    }
  }

  const int mb = qblk * 128 + wave * 16 + ((lane >> 4) << 3);
  const int nc = lane & 15;
#pragma unroll
  for (int nf = 0; nf < 4; ++nf)
#pragma unroll
    for (int r = 0; r < 8; ++r)
      O[base + (size_t)(mb + r) * 512 + nf * 16 + nc] = oacc[nf][r];
}

// ---------------------------------------------------------------------------
// BatchNorm stats (two deterministic passes over 16384 rows, per column)
// ---------------------------------------------------------------------------
__global__ __launch_bounds__(256) void stats_part_kernel(
    const float* __restrict__ Hm, float* __restrict__ ps, float* __restrict__ ps2, int D) {
  int gid = blockIdx.x * 256 + threadIdx.x;  // [0, 16*D)
  int c = gid % D, chunk = gid / D;
  float s = 0.f, s2 = 0.f;
  int r0 = chunk * 1024;
  for (int r = 0; r < 1024; ++r) {
    float v = Hm[(size_t)(r0 + r) * D + c];
    s += v; s2 += v * v;
  }
  ps[gid] = s; ps2[gid] = s2;
}

__global__ __launch_bounds__(256) void stats_final_kernel(
    const float* __restrict__ ps, const float* __restrict__ ps2,
    float* __restrict__ mean, float* __restrict__ rstd, int D) {
  int c = blockIdx.x * 256 + threadIdx.x;
  if (c >= D) return;
  float s = 0.f, s2 = 0.f;
  for (int i = 0; i < 16; ++i) { s += ps[i * D + c]; s2 += ps2[i * D + c]; }
  float m = s * (1.f / 16384.f);
  float var = s2 * (1.f / 16384.f) - m * m;  // biased, matching jnp.var
  mean[c] = m;
  rstd[c] = rsqrtf(var + 1e-5f);
}

// ---------------------------------------------------------------------------
// BN (optional) + LIF scan over T=4 (TAU=2, THR=1, detach-reset).
// Hm layout (T, B*N, D). vt=1: write spikes transposed per head,
// layout (T,B,H,hd,N), for the attention v operand.
// ---------------------------------------------------------------------------
__global__ __launch_bounds__(256) void bn_lif_kernel(
    const float* __restrict__ Hm, const float* __restrict__ mean,
    const float* __restrict__ rstd, const float* __restrict__ gamma,
    const float* __restrict__ beta, __bf16* __restrict__ spk,
    int D, int use_bn, int vt) {
  size_t per_t = (size_t)4096 * D;
  size_t idx = (size_t)blockIdx.x * 256 + threadIdx.x;
  if (idx >= per_t) return;
  int c = (int)(idx % D);
  int r = (int)(idx / D);
  float mu = 0.f, rs = 1.f, ga = 1.f, be = 0.f;
  if (use_bn) { mu = mean[c]; rs = rstd[c]; ga = gamma[c]; be = beta[c]; }
  float mem = 0.f;
#pragma unroll
  for (int t = 0; t < 4; ++t) {
    float v = (Hm[(size_t)t * per_t + idx] - mu) * rs * ga + be;
    mem = mem + (v - mem) * 0.5f;               // TAU = 2
    float s = (mem - 1.0f) > 0.f ? 1.f : 0.f;   // THR = 1
    size_t oi;
    if (vt) {  // (t, b, h, d, n)
      int b_ = r >> 8, n_ = r & 255, h_ = c >> 6, d_ = c & 63;
      oi = ((((size_t)t * 16 + b_) * 8 + h_) * 64 + d_) * 256 + n_;
    } else {
      oi = (size_t)t * per_t + idx;
    }
    spk[oi] = (__bf16)s;
    mem = mem * (1.f - s);
  }
}

// ---------------------------------------------------------------------------
// y = x + spikes ; optional bf16 copy of y for the next GEMM
// ---------------------------------------------------------------------------
__global__ __launch_bounds__(256) void add_res_kernel(
    const float* __restrict__ x, const __bf16* __restrict__ s,
    float* __restrict__ y, __bf16* __restrict__ yb, size_t n) {
  size_t i = (size_t)blockIdx.x * 256 + threadIdx.x;
  if (i >= n) return;
  float v = x[i] + (float)s[i];
  y[i] = v;
  if (yb) yb[i] = (__bf16)v;
}

// ---------------------------------------------------------------------------
extern "C" void kernel_launch(void* const* d_in, const int* in_sizes, int n_in,
                              void* d_out, int out_size, void* d_ws, size_t ws_size,
                              hipStream_t stream) {
  const float* x    = (const float*)d_in[0];
  const float* qW   = (const float*)d_in[1],  *qb  = (const float*)d_in[2];
  const float* qg   = (const float*)d_in[3],  *qbe = (const float*)d_in[4];
  const float* kW   = (const float*)d_in[5],  *kb_ = (const float*)d_in[6];
  const float* kg   = (const float*)d_in[7],  *kbe = (const float*)d_in[8];
  const float* vW   = (const float*)d_in[9],  *vb  = (const float*)d_in[10];
  const float* vg   = (const float*)d_in[11], *vbe = (const float*)d_in[12];
  const float* pW   = (const float*)d_in[13], *pb  = (const float*)d_in[14];
  const float* pg   = (const float*)d_in[15], *pbe = (const float*)d_in[16];
  const float* f1W  = (const float*)d_in[17], *f1b = (const float*)d_in[18];
  const float* f1g  = (const float*)d_in[19], *f1be = (const float*)d_in[20];
  const float* f2W  = (const float*)d_in[21], *f2b = (const float*)d_in[22];
  const float* f2g  = (const float*)d_in[23], *f2be = (const float*)d_in[24];

  const size_t XB = (size_t)16384 * 512;  // one (T,B,N,C) tensor

  char* w = (char*)d_ws;
  auto carve = [&](size_t bytes) -> void* {
    void* p = (void*)w;
    w += (bytes + 255) & ~(size_t)255;
    return p;
  };
  __bf16* wqb  = (__bf16*)carve(262144 * 2);   // all weights stored transposed
  __bf16* wkb  = (__bf16*)carve(262144 * 2);
  __bf16* wvb  = (__bf16*)carve(262144 * 2);
  __bf16* wpb  = (__bf16*)carve(262144 * 2);
  __bf16* wf1b = (__bf16*)carve(1048576 * 2);
  __bf16* wf2b = (__bf16*)carve(1048576 * 2);
  __bf16* xb   = (__bf16*)carve(XB * 2);
  __bf16* x1b  = (__bf16*)carve(XB * 2);
  __bf16* spk0 = (__bf16*)carve(4 * XB * 2);   // sq|sk|svT|so ; later sf1/sf2
  float*  x1   = (float*)carve(XB * 4);
  float*  h    = (float*)carve((size_t)16384 * 2048 * 4);
  float*  mean = (float*)carve(2048 * 4);
  float*  rstd = (float*)carve(2048 * 4);
  float*  ps   = (float*)carve(16 * 2048 * 4);
  float*  ps2  = (float*)carve(16 * 2048 * 4);

  __bf16* sq  = spk0;
  __bf16* sk  = spk0 + XB;
  __bf16* svT = spk0 + 2 * XB;
  __bf16* so  = spk0 + 3 * XB;
  __bf16* sf1 = spk0;  // 4*XB == 16384x2048 bf16, overlays dead sq..so
  __bf16* sf2 = spk0;  // 1*XB, after sf1 is dead

  auto cvt = [&](const float* src, __bf16* dst, size_t n) {
    cvt_bf16_kernel<<<dim3((unsigned)(n / 256)), 256, 0, stream>>>(src, dst, n);
  };
  auto cvtT = [&](const float* src, __bf16* dst, int K, int N) {
    cvt_wt_kernel<<<dim3((unsigned)((size_t)K * N / 256)), 256, 0, stream>>>(src, dst, K, N);
  };
  auto gemm = [&](const __bf16* A, const __bf16* Wt, const float* bias, float* C,
                  int K, int Nd) {
    gemm_bf16_kernel<<<dim3(16384 / 128, Nd / 64), 256, 0, stream>>>(A, Wt, bias, C,
                                                                     16384, K, Nd);
  };
  auto stats = [&](const float* Hm, int D) {
    stats_part_kernel<<<dim3(D * 16 / 256), 256, 0, stream>>>(Hm, ps, ps2, D);
    stats_final_kernel<<<dim3(D / 256), 256, 0, stream>>>(ps, ps2, mean, rstd, D);
  };
  auto bnlif = [&](const float* Hm, const float* g, const float* be, __bf16* spk,
                   int D, int use_bn, int vt) {
    bn_lif_kernel<<<dim3((unsigned)((size_t)4096 * D / 256)), 256, 0, stream>>>(
        Hm, mean, rstd, g, be, spk, D, use_bn, vt);
  };

  // bf16 packs (weights transposed to N-major for K-contiguous B tiles)
  cvt(x, xb, XB);
  cvtT(qW, wqb, 512, 512);   cvtT(kW, wkb, 512, 512);
  cvtT(vW, wvb, 512, 512);   cvtT(pW, wpb, 512, 512);
  cvtT(f1W, wf1b, 512, 2048);
  cvtT(f2W, wf2b, 2048, 512);

  // q / k / v branches: GEMM -> BN stats -> BN+LIF spikes
  gemm(xb, wqb, qb, h, 512, 512);  stats(h, 512);  bnlif(h, qg, qbe, sq, 512, 1, 0);
  gemm(xb, wkb, kb_, h, 512, 512); stats(h, 512);  bnlif(h, kg, kbe, sk, 512, 1, 0);
  gemm(xb, wvb, vb, h, 512, 512);  stats(h, 512);  bnlif(h, vg, vbe, svT, 512, 1, 1);

  // fused spiking attention (o written into h), then plain LIF
  attn_kernel<<<dim3(1024), 256, 0, stream>>>(sq, sk, svT, h);
  bnlif(h, nullptr, nullptr, so, 512, 0, 0);

  // projection + residual (proj spikes reuse the dead sk slot)
  gemm(so, wpb, pb, h, 512, 512);  stats(h, 512);  bnlif(h, pg, pbe, sk, 512, 1, 0);
  add_res_kernel<<<dim3((unsigned)(XB / 256)), 256, 0, stream>>>(x, sk, x1, x1b, XB);

  // MLP
  gemm(x1b, wf1b, f1b, h, 512, 2048); stats(h, 2048); bnlif(h, f1g, f1be, sf1, 2048, 1, 0);
  gemm(sf1, wf2b, f2b, h, 2048, 512); stats(h, 512);  bnlif(h, f2g, f2be, sf2, 512, 1, 0);
  add_res_kernel<<<dim3((unsigned)(XB / 256)), 256, 0, stream>>>(x1, sf2, (float*)d_out,
                                                                 nullptr, XB);
}
